// GIN_67147518705672
// MI455X (gfx1250) — matrix operand order repaired and checked by reading, not compiled, not run on hardware
//
#include <hip/hip_runtime.h>

#define N_NODES 8192
#define DIM     256

typedef __attribute__((ext_vector_type(16))) __bf16       v16bf;
typedef __attribute__((ext_vector_type(2)))  __bf16       v2bf;
typedef __attribute__((ext_vector_type(8)))  float        v8f;
typedef __attribute__((ext_vector_type(8)))  unsigned int v8u;

// LDS tile strides in ushorts. 40 ushorts = 80 B: 16-B aligned rows (b128
// async stores) and fragment lane stride of 20 dwords -> 16 distinct banks.
#define A_LD 40
#define B_LD 40
#define AS_BYTES (64 * A_LD * 2)    // 5120 B per buffer
#define BS_BYTES (256 * B_LD * 2)   // 20480 B per buffer

// Native bf16 conversions -> v_cvt_pk_bf16_f32 (RNE).
__device__ __forceinline__ unsigned short f2bf(float a) {
    __bf16 h = (__bf16)a;
    return __builtin_bit_cast(unsigned short, h);
}
__device__ __forceinline__ unsigned int pack_bf2(float a, float b) {
    v2bf p;
    p[0] = (__bf16)a;
    p[1] = (__bf16)b;
    return __builtin_bit_cast(unsigned int, p);
}

// Low 32 bits of a flat shared pointer == LDS byte offset (ISA 10.2 aperture).
__device__ __forceinline__ unsigned int lds_off32(const void* p) {
    return (unsigned int)(size_t)p;
}

// CDNA5 async global->LDS copy, 16 B per lane, tracked by ASYNCcnt.
__device__ __forceinline__ void async_copy_b128(unsigned int lds_byte_off,
                                                unsigned int glob_byte_off,
                                                const void* sbase) {
    asm volatile("global_load_async_to_lds_b128 %0, %1, %2 offset:0"
                 :
                 : "v"(lds_byte_off), "v"(glob_byte_off), "s"(sbase)
                 : "memory");
}
__device__ __forceinline__ void wait_async0() {
    asm volatile("s_wait_asynccnt 0" ::: "memory");
}

// 16x32 bf16 A-fragment (two ds_load_b128 per lane).
__device__ __forceinline__ v16bf load_a_frag(const unsigned short* As, int mw,
                                             int l16, int half) {
    const int mrow = mw * 16 + l16;
    v8u au;
#pragma unroll
    for (int r = 0; r < 8; ++r) {
        const int k = (r < 4) ? (2 * r + 8 * half) : (2 * r + 8 + 8 * half);
        au[r] = *(const unsigned int*)(As + mrow * A_LD + k);
    }
    return __builtin_bit_cast(v16bf, au);
}

// 32x16 bf16 B-fragment from B^T tile.
__device__ __forceinline__ v16bf load_b_frag(const unsigned short* Bst, int nb,
                                             int l16, int half) {
    const int n = nb + l16;
    v8u bu;
#pragma unroll
    for (int r = 0; r < 8; ++r) {
        const int k = (r < 4) ? (2 * r + 8 * half) : (2 * r + 8 + 8 * half);
        bu[r] = *(const unsigned int*)(Bst + n * B_LD + k);
    }
    return __builtin_bit_cast(v16bf, bu);
}

// ---------------------------------------------------------------------------
// Pre-pass: xT[n][m] = bf16(x[m][n]).  LDS 64x64 tile transpose.
// ---------------------------------------------------------------------------
__global__ __launch_bounds__(256) void transpose_cvt(const float* __restrict__ x,
                                                     unsigned short* __restrict__ xT) {
    __shared__ unsigned short Tl[64 * 66];
    const int t  = threadIdx.x;
    const int m0 = blockIdx.x * 64;
    const int n0 = blockIdx.y * 64;
#pragma unroll
    for (int j = 0; j < 4; ++j) {
        const int idx = t + j * 256;
        const int m   = idx >> 4;
        const int c   = idx & 15;
        const float4 v = *(const float4*)(x + (size_t)(m0 + m) * DIM + n0 + c * 4);
        Tl[(c * 4 + 0) * 66 + m] = f2bf(v.x);
        Tl[(c * 4 + 1) * 66 + m] = f2bf(v.y);
        Tl[(c * 4 + 2) * 66 + m] = f2bf(v.z);
        Tl[(c * 4 + 3) * 66 + m] = f2bf(v.w);
    }
    __syncthreads();
#pragma unroll
    for (int j = 0; j < 8; ++j) {
        const int idx = t + j * 256;
        const int n   = idx >> 5;
        const int c   = idx & 31;
        const unsigned int v = *(const unsigned int*)(Tl + n * 66 + c * 2);
        *((unsigned int*)(xT + (size_t)(n0 + n) * N_NODES + m0) + c) = v;
    }
}

// Pre-pass: Wbf = bf16(W).
__global__ __launch_bounds__(256) void cvt_w(const float* __restrict__ W,
                                             unsigned short* __restrict__ Wbf) {
    const int idx = blockIdx.x * 256 + threadIdx.x;
    const float4 v = *(const float4*)(W + (size_t)idx * 4);
    unsigned int* dst = (unsigned int*)Wbf + idx * 2;
    dst[0] = pack_bf2(v.x, v.y);
    dst[1] = pack_bf2(v.z, v.w);
}

// ---------------------------------------------------------------------------
// ybf = bf16( xin + eps * adj @ x ).  adj streamed fp32->bf16 in registers;
// x^T tile fetched by async global->LDS b128 copies into double-buffered LDS.
// ---------------------------------------------------------------------------
__global__ __launch_bounds__(256) void gin_agg(const float* __restrict__ adj,
                                               const float* __restrict__ xin,
                                               const unsigned short* __restrict__ xT,
                                               const float* __restrict__ eps_ptr,
                                               unsigned short* __restrict__ ybf) {
    __shared__ unsigned short As[2][64 * A_LD];
    __shared__ unsigned short Bst[2][256 * B_LD];

    const int t    = threadIdx.x;
    const int lane = t & 31;
    const int wave = t >> 5;
    const int mw   = wave >> 1;
    const int nw   = wave & 1;
    const int half = lane >> 4;
    const int l16  = lane & 15;
    const int blockM = blockIdx.x * 64;

    const int r0 = t >> 3, ac = t & 7;   // adj staging: rows r0, r0+32
    const int nB = t >> 2, cB = t & 3;   // x^T staging: rows nB+64j, 16B chunk cB

    v8f acc[8];
#pragma unroll
    for (int f = 0; f < 8; ++f)
#pragma unroll
        for (int r = 0; r < 8; ++r) acc[f][r] = 0.0f;

    // async copy descriptors for the B tile
    const unsigned int bst0 = lds_off32(&Bst[0][0]);
    unsigned int bl[4], bg[4];
#pragma unroll
    for (int j = 0; j < 4; ++j) {
        const int n = nB + j * 64;
        bl[j] = bst0 + (unsigned)n * (B_LD * 2) + cB * 16;
        bg[j] = ((unsigned)n * N_NODES + cB * 8) * 2;
    }
    // prologue: tile 0 in flight
#pragma unroll
    for (int j = 0; j < 4; ++j) { async_copy_b128(bl[j], bg[j], xT); bg[j] += 64; }

    const float* aP0 = adj + (size_t)(blockM + r0) * N_NODES + ac * 4;
    const float* aP1 = aP0 + (size_t)32 * N_NODES;
    float4 aA = *(const float4*)(aP0);
    float4 aB = *(const float4*)(aP1);

    for (int k0 = 0; k0 < N_NODES; k0 += 32) {
        const int c = (k0 >> 5) & 1;
        // commit adj registers -> As[c] (v_cvt_pk_bf16_f32 + ds_store)
        unsigned int* d0 = (unsigned int*)As[c] + r0 * (A_LD / 2) + ac * 2;
        d0[0] = pack_bf2(aA.x, aA.y);
        d0[1] = pack_bf2(aA.z, aA.w);
        unsigned int* d1 = (unsigned int*)As[c] + (r0 + 32) * (A_LD / 2) + ac * 2;
        d1[0] = pack_bf2(aB.x, aB.y);
        d1[1] = pack_bf2(aB.z, aB.w);

        wait_async0();        // our Bst[c] async copies are complete
        __syncthreads();      // everyone's As[c]/Bst[c] visible

        if (k0 + 32 < N_NODES) {
            // next B tile -> other buffer (overlaps WMMAs, ASYNCcnt-tracked)
#pragma unroll
            for (int j = 0; j < 4; ++j) {
                async_copy_b128(bl[j] + (unsigned)(1 - c) * BS_BYTES, bg[j], xT);
                bg[j] += 64;
            }
            // next adj tile -> registers (waits land at next iter's convert)
            aA = *(const float4*)(aP0 + k0 + 32);
            aB = *(const float4*)(aP1 + k0 + 32);
        }

        const v16bf a = load_a_frag(As[c], mw, l16, half);
#pragma unroll
        for (int f = 0; f < 8; ++f) {
            const v16bf b = load_b_frag(Bst[c], nw * 128 + f * 16, l16, half);
            acc[f] = __builtin_amdgcn_wmma_f32_16x16x32_bf16(
                false, a, false, b, (short)0, acc[f], false, false);
        }
    }

    const float eps = eps_ptr[0];
#pragma unroll
    for (int f = 0; f < 8; ++f) {
        const int gn = nw * 128 + f * 16 + l16;
#pragma unroll
        for (int r = 0; r < 8; ++r) {
            const int gm = blockM + mw * 16 + r + 8 * half;
            const size_t o = (size_t)gm * DIM + gn;
            ybf[o] = f2bf(xin[o] + eps * acc[f][r]);
        }
    }
}

// ---------------------------------------------------------------------------
// out = act(in @ W^T + b).  Both tiles bf16 -> fully async staging.
// ---------------------------------------------------------------------------
template <int RELU, int OUTBF>
__global__ __launch_bounds__(256) void gin_mlp(const unsigned short* __restrict__ inb,
                                               const unsigned short* __restrict__ Wbf,
                                               const float* __restrict__ bias,
                                               void* __restrict__ outp) {
    __shared__ unsigned short As[2][64 * A_LD];
    __shared__ unsigned short Bst[2][256 * B_LD];

    const int t    = threadIdx.x;
    const int lane = t & 31;
    const int wave = t >> 5;
    const int mw   = wave >> 1;
    const int nw   = wave & 1;
    const int half = lane >> 4;
    const int l16  = lane & 15;
    const int blockM = blockIdx.x * 64;
    const int rA = t >> 2, cA = t & 3;

    v8f acc[8];
#pragma unroll
    for (int f = 0; f < 8; ++f)
#pragma unroll
        for (int r = 0; r < 8; ++r) acc[f][r] = 0.0f;

    const unsigned int as0  = lds_off32(&As[0][0]);
    const unsigned int bst0 = lds_off32(&Bst[0][0]);
    const unsigned int al = as0 + rA * (A_LD * 2) + cA * 16;
    const unsigned int ag = ((unsigned)(blockM + rA) * DIM + cA * 8) * 2;
    const unsigned int blb = bst0 + rA * (B_LD * 2) + cA * 16;
    const unsigned int bgb = ((unsigned)rA * DIM + cA * 8) * 2;

    // prologue: tile 0 in flight
    async_copy_b128(al, ag, inb);
#pragma unroll
    for (int j = 0; j < 4; ++j)
        async_copy_b128(blb + (unsigned)j * 64 * (B_LD * 2),
                        bgb + (unsigned)j * 64 * (DIM * 2), Wbf);

#pragma unroll 1
    for (int k0 = 0; k0 < DIM; k0 += 32) {
        const int c = (k0 >> 5) & 1;
        wait_async0();
        __syncthreads();
        if (k0 + 32 < DIM) {
            const unsigned int koff = (unsigned)(k0 + 32) * 2;
            async_copy_b128(al + (unsigned)(1 - c) * AS_BYTES, ag + koff, inb);
#pragma unroll
            for (int j = 0; j < 4; ++j)
                async_copy_b128(blb + (unsigned)(1 - c) * BS_BYTES + (unsigned)j * 64 * (B_LD * 2),
                                bgb + (unsigned)j * 64 * (DIM * 2) + koff, Wbf);
        }
        const v16bf a = load_a_frag(As[c], mw, l16, half);
#pragma unroll
        for (int f = 0; f < 8; ++f) {
            const v16bf b = load_b_frag(Bst[c], nw * 128 + f * 16, l16, half);
            acc[f] = __builtin_amdgcn_wmma_f32_16x16x32_bf16(
                false, a, false, b, (short)0, acc[f], false, false);
        }
    }

#pragma unroll
    for (int f = 0; f < 8; ++f) {
        const int gn = nw * 128 + f * 16 + l16;
        const float bn = bias[gn];
#pragma unroll
        for (int r = 0; r < 8; ++r) {
            const int gm = blockM + mw * 16 + r + 8 * half;
            float v = acc[f][r] + bn;
            if (RELU) v = fmaxf(v, 0.0f);
            const size_t o = (size_t)gm * DIM + gn;
            if (OUTBF) ((unsigned short*)outp)[o] = f2bf(v);
            else       ((float*)outp)[o] = v;
        }
    }
}

extern "C" void kernel_launch(void* const* d_in, const int* in_sizes, int n_in,
                              void* d_out, int out_size, void* d_ws, size_t ws_size,
                              hipStream_t stream) {
    (void)in_sizes; (void)n_in; (void)out_size; (void)ws_size;
    const float* x   = (const float*)d_in[0];
    const float* adj = (const float*)d_in[1];
    const float* W1  = (const float*)d_in[2];
    const float* b1  = (const float*)d_in[3];
    const float* W2  = (const float*)d_in[4];
    const float* b2  = (const float*)d_in[5];
    const float* eps = (const float*)d_in[6];
    float* out = (float*)d_out;

    const size_t feat = (size_t)N_NODES * DIM;
    char* w = (char*)d_ws;
    unsigned short* xTbf = (unsigned short*)w;  w += feat * 2;
    unsigned short* ybf  = (unsigned short*)w;  w += feat * 2;
    unsigned short* hbf  = (unsigned short*)w;  w += feat * 2;
    float*          x1   = (float*)w;           w += feat * 4;
    unsigned short* W1bf = (unsigned short*)w;  w += (size_t)DIM * DIM * 2;
    unsigned short* W2bf = (unsigned short*)w;  w += (size_t)DIM * DIM * 2;

    dim3 block(256);
    cvt_w<<<dim3(DIM * DIM / 1024), block, 0, stream>>>(W1, W1bf);
    cvt_w<<<dim3(DIM * DIM / 1024), block, 0, stream>>>(W2, W2bf);

    const float* cur = x;
    for (int layer = 0; layer < 2; ++layer) {
        transpose_cvt<<<dim3(N_NODES / 64, DIM / 64), block, 0, stream>>>(cur, xTbf);
        gin_agg<<<dim3(N_NODES / 64), block, 0, stream>>>(adj, cur, xTbf, eps, ybf);
        gin_mlp<1, 1><<<dim3(N_NODES / 64), block, 0, stream>>>(ybf, W1bf, b1, hbf);
        void* dst = (layer == 0) ? (void*)x1 : (void*)out;
        gin_mlp<0, 0><<<dim3(N_NODES / 64), block, 0, stream>>>(hbf, W2bf, b2, dst);
        cur = x1;
    }
}
